// AoA_Decoder_26585847562868
// MI455X (gfx1250) — matrix operand order repaired and checked by reading, not compiled, not run on hardware
//
#include <hip/hip_runtime.h>
#include <hip/hip_bf16.h>

// Problem constants (match reference)
#define B_   128
#define N_   36
#define D_   1024
#define H_   1024
#define E_   1024
#define V_   10000
#define VPAD 10240   // vocab rows padded to a multiple of the 512-col block tile
#define T_   20
#define NH_  8
#define DH_  128
#define KCOMB 3072   // E + H (xin) + H (hidden) for fused gates GEMM

typedef __bf16 bf16_t;
typedef bf16_t v16bf __attribute__((ext_vector_type(16)));
typedef bf16_t v8bf  __attribute__((ext_vector_type(8)));
typedef float  v8f   __attribute__((ext_vector_type(8)));

__device__ __forceinline__ float sigmoidf_(float x) { return 1.0f / (1.0f + __expf(-x)); }

// ---------------------------------------------------------------------------
// Generic bf16 WMMA GEMM:  C[M,N] = A[M,K] @ W[N,K]^T + bias[n]
// Block = 256 threads = 8 waves arranged 2(M) x 4(N); wave tile = 16M x 128N
// (8 accumulators), block tile = 32M x 512N.  N (padded) must be a multiple
// of 512 and M a multiple of 32.  Per K-step: all 8 B-fragments are loaded
// into DISTINCT registers first (one load clause, loads stay in flight),
// then 8 WMMAs issue with partial loadcnt waits -- load/math overlap.
// Store epilogue masks col < Nreal and (optionally) the sequence-length mask.
// A-fragment (16x32 bf16): lane=16*half+m holds K = half*16..+15 of row m
// (contiguous). B-fragment analogous to the documented sparse B
// layout: lane=16*half+n holds K = half*16..+15 of W row n (contiguous).
// ---------------------------------------------------------------------------
__global__ __launch_bounds__(256) void k_wmma_gemm_nt(
    const bf16_t* __restrict__ A, int lda,
    const bf16_t* __restrict__ W, int ldw,
    const float* __restrict__ bias,
    float* __restrict__ C, int ldc,
    int Nreal, int K,
    const int* __restrict__ lengths, int tstep)
{
    const int tid  = threadIdx.x;
    const int wave = tid >> 5;
    const int lane = tid & 31;
    const int half = lane >> 4;
    const int lr   = lane & 15;
    const int m0 = blockIdx.y * 32 + (wave & 1) * 16;
    const int n0 = blockIdx.x * 512 + (wave >> 1) * 128;

    const bf16_t* arow  = A + (size_t)(m0 + lr) * lda + half * 16;
    const bf16_t* wbase = W + (size_t)(n0 + lr) * ldw + half * 16;
    const size_t  wtile = (size_t)16 * ldw;   // row jump between column tiles

    v8f acc[8] = {};

    for (int k0 = 0; k0 < K; k0 += 32) {
        // ---- load phase: A fragment + all 8 B fragments, distinct registers ----
        v8bf alo = *(const v8bf*)(arow + k0);
        v8bf ahi = *(const v8bf*)(arow + k0 + 8);
        v16bf bfrag[8];
#pragma unroll
        for (int c = 0; c < 8; ++c)
            bfrag[c] = *(const v16bf*)(wbase + (size_t)c * wtile + k0);
        if (k0 + 256 < K) {
            __builtin_prefetch(wbase + k0 + 256, 0, 1);             // global_prefetch_b8
            __builtin_prefetch(wbase + 4 * wtile + k0 + 256, 0, 1);
        }
        v16bf afrag = __builtin_shufflevector(alo, ahi,
            0, 1, 2, 3, 4, 5, 6, 7, 8, 9, 10, 11, 12, 13, 14, 15);
        // ---- math phase: 8 WMMAs; later B loads still in flight ----
#pragma unroll
        for (int c = 0; c < 8; ++c)
            acc[c] = __builtin_amdgcn_wmma_f32_16x16x32_bf16(
                false, afrag, false, bfrag[c], (short)0, acc[c], false, false);
    }

#pragma unroll
    for (int c = 0; c < 8; ++c) {
        int col = n0 + c * 16 + lr;
        if (col >= Nreal) continue;
        float bb = bias ? bias[col] : 0.0f;
#pragma unroll
        for (int j = 0; j < 8; ++j) {
            int row = m0 + half * 8 + j;      // C layout: vgpr j -> M = 8*half + j
            float v = acc[c][j] + bb;
            if (lengths) v = (tstep < lengths[row]) ? v : 0.0f;
            C[(size_t)row * ldc + col] = v;
        }
    }
}

// ---------------------------------------------------------------------------
// Prep kernels (fp32 -> bf16 weight conversion, packing, weight-norm)
// ---------------------------------------------------------------------------
__global__ __launch_bounds__(256) void k_cvt_bf16(const float* __restrict__ in,
                                                  bf16_t* __restrict__ out, int n)
{
    int i = blockIdx.x * 256 + threadIdx.x;
    int stride = gridDim.x * 256;
    for (; i < n; i += stride) out[i] = (bf16_t)in[i];
}

__global__ __launch_bounds__(256) void k_pack_comb(const float* __restrict__ Wih,
                                                   const float* __restrict__ Whh,
                                                   bf16_t* __restrict__ out)
{
    int i = blockIdx.x * 256 + threadIdx.x;
    const int total = 4 * H_ * KCOMB;
    int stride = gridDim.x * 256;
    for (; i < total; i += stride) {
        int r = i / KCOMB, c = i - r * KCOMB;
        float v = (c < (E_ + H_)) ? Wih[(size_t)r * (E_ + H_) + c]
                                  : Whh[(size_t)r * H_ + (c - (E_ + H_))];
        out[i] = (bf16_t)v;
    }
}

__global__ __launch_bounds__(256) void k_bias_sum(const float* __restrict__ a,
                                                  const float* __restrict__ b,
                                                  float* __restrict__ out, int n)
{
    int i = blockIdx.x * 256 + threadIdx.x;
    if (i < n) out[i] = a[i] + b[i];
}

// Wpred = g * V / ||V||_row, converted to bf16.  One block per vocab row.
// Pad rows [V_, VPAD) are zeroed by a memset before this runs.
__global__ __launch_bounds__(256) void k_wnorm(const float* __restrict__ Vv,
                                               const float* __restrict__ g,
                                               bf16_t* __restrict__ out)
{
    int r = blockIdx.x;
    __shared__ float red[256];
    const float* row = Vv + (size_t)r * H_;
    float s = 0.0f;
    for (int j = threadIdx.x; j < H_; j += 256) { float v = row[j]; s += v * v; }
    red[threadIdx.x] = s;
    __syncthreads();
    for (int st = 128; st > 0; st >>= 1) {
        if (threadIdx.x < st) red[threadIdx.x] += red[threadIdx.x + st];
        __syncthreads();
    }
    float scale = g[r] * rsqrtf(red[0] + 1e-30f);
    for (int j = threadIdx.x; j < H_; j += 256)
        out[(size_t)r * H_ + j] = (bf16_t)(row[j] * scale);
}

__global__ __launch_bounds__(256) void k_meanfeat(const float* __restrict__ enc,
                                                  float* __restrict__ mf)
{
    int i = blockIdx.x * 256 + threadIdx.x;
    if (i >= B_ * D_) return;
    int b = i >> 10, d = i & (D_ - 1);
    float s = 0.0f;
    for (int n = 0; n < N_; ++n) s += enc[((size_t)b * N_ + n) * D_ + d];
    mf[i] = s * (1.0f / N_);
}

// ---------------------------------------------------------------------------
// Per-step pointwise kernels
// ---------------------------------------------------------------------------
// xin[b, 0:1024]   = relu(emb_W[tok])
// xin[b, 1024:2048]= mean_feat + ctx          (cols 2048:3072 = h, kept from lstm)
__global__ __launch_bounds__(256) void k_embed_concat(
    const int* __restrict__ caps, const float* __restrict__ embW,
    const float* __restrict__ mf, const float* __restrict__ ctx,
    bf16_t* __restrict__ xin, int t)
{
    int i = blockIdx.x * 256 + threadIdx.x;   // B * 2048
    if (i >= B_ * 2048) return;
    int b = i >> 11, c = i & 2047;
    float v;
    if (c < E_) {
        int tok = caps[b * T_ + t];
        v = fmaxf(embW[(size_t)tok * E_ + c], 0.0f);
    } else {
        int j = c - E_;
        v = mf[b * D_ + j] + ctx[b * H_ + j];
    }
    xin[(size_t)b * KCOMB + c] = (bf16_t)v;
}

// LSTM cell + layernorm (ddof=1, eps on std). One block per batch row.
// Writes: m (fp32, in/out), h as bf16 into xin[:,2048:3072], q as bf16 to q_bf
// and into zin[:,1024:2048].
__global__ __launch_bounds__(256) void k_lstm_ln(
    const float* __restrict__ gates, float* __restrict__ m,
    const float* __restrict__ lng, const float* __restrict__ lnb,
    bf16_t* __restrict__ xin, bf16_t* __restrict__ q_bf, bf16_t* __restrict__ zin)
{
    int b = blockIdx.x;
    int tid = threadIdx.x;
    __shared__ float sh[H_];
    __shared__ float rsum[256], rsq[256];
    const float* g = gates + (size_t)b * 4 * H_;
    float lsum = 0.0f, lsq = 0.0f;
#pragma unroll
    for (int it = 0; it < H_ / 256; ++it) {
        int j = tid + it * 256;
        float ig = g[j], fg = g[H_ + j], gg = g[2 * H_ + j], og = g[3 * H_ + j];
        float mo = m[b * H_ + j];
        float m2 = sigmoidf_(fg) * mo + sigmoidf_(ig) * tanhf(gg);
        float h2 = sigmoidf_(og) * tanhf(m2);
        m[b * H_ + j] = m2;
        xin[(size_t)b * KCOMB + (E_ + H_) + j] = (bf16_t)h2;
        sh[j] = h2;
        lsum += h2; lsq += h2 * h2;
    }
    rsum[tid] = lsum; rsq[tid] = lsq;
    __syncthreads();
    for (int st = 128; st > 0; st >>= 1) {
        if (tid < st) { rsum[tid] += rsum[tid + st]; rsq[tid] += rsq[tid + st]; }
        __syncthreads();
    }
    float mu  = rsum[0] * (1.0f / H_);
    float var = (rsq[0] - (float)H_ * mu * mu) * (1.0f / (H_ - 1));
    float inv = 1.0f / (sqrtf(fmaxf(var, 0.0f)) + 1e-6f);
#pragma unroll
    for (int it = 0; it < H_ / 256; ++it) {
        int j = tid + it * 256;
        float qv = lng[j] * (sh[j] - mu) * inv + lnb[j];
        q_bf[b * H_ + j] = (bf16_t)qv;
        zin[(size_t)b * 2 * D_ + D_ + j] = (bf16_t)qv;
    }
}

// Multi-head attention, N=36 (tiny): one thread per (b, head).
// Writes att into zin[:, 0:1024] (bf16) for the AoA GEMM.
__global__ __launch_bounds__(256) void k_attention(
    const float* __restrict__ qp, const float* __restrict__ kp,
    const float* __restrict__ vp, bf16_t* __restrict__ zin)
{
    int idx = blockIdx.x * 256 + threadIdx.x;   // B * NH
    if (idx >= B_ * NH_) return;
    int b = idx >> 3, h = idx & (NH_ - 1);
    const float* q = qp + (size_t)b * D_ + h * DH_;
    const float scale = 0.08838834764831843f;   // 1/sqrt(128)
    float s[N_];
    float mx = -1e30f;
#pragma unroll
    for (int n = 0; n < N_; ++n) {
        const float* kr = kp + ((size_t)b * N_ + n) * D_ + h * DH_;
        float d = 0.0f;
#pragma unroll
        for (int dd = 0; dd < DH_; ++dd) d += q[dd] * kr[dd];
        s[n] = d * scale;
        mx = fmaxf(mx, s[n]);
    }
    float denom = 0.0f;
#pragma unroll
    for (int n = 0; n < N_; ++n) { s[n] = __expf(s[n] - mx); denom += s[n]; }
    float inv = 1.0f / denom;
    for (int dd = 0; dd < DH_; ++dd) {
        float a = 0.0f;
#pragma unroll
        for (int n = 0; n < N_; ++n)
            a += s[n] * vp[((size_t)b * N_ + n) * D_ + h * DH_ + dd];
        zin[(size_t)b * 2 * D_ + h * DH_ + dd] = (bf16_t)(a * inv);
    }
}

// GLU:  ctx = z[:, :1024] * sigmoid(z[:, 1024:])
__global__ __launch_bounds__(256) void k_glu(const float* __restrict__ z,
                                             float* __restrict__ ctx,
                                             bf16_t* __restrict__ ctx_bf)
{
    int i = blockIdx.x * 256 + threadIdx.x;   // B * H
    if (i >= B_ * H_) return;
    int b = i >> 10, j = i & (H_ - 1);
    float a  = z[(size_t)b * 2 * D_ + j];
    float gt = z[(size_t)b * 2 * D_ + D_ + j];
    float c = a * sigmoidf_(gt);
    ctx[i] = c;
    ctx_bf[i] = (bf16_t)c;
}

// ---------------------------------------------------------------------------
// Host launcher
// ---------------------------------------------------------------------------
extern "C" void kernel_launch(void* const* d_in, const int* in_sizes, int n_in,
                              void* d_out, int out_size, void* d_ws, size_t ws_size,
                              hipStream_t stream)
{
    (void)in_sizes; (void)n_in; (void)out_size; (void)ws_size;

    const float* enc   = (const float*)d_in[0];
    const int*   caps  = (const int*)d_in[1];
    const int*   lens  = (const int*)d_in[2];
    const float* embW  = (const float*)d_in[3];
    const float* Wih   = (const float*)d_in[4];
    const float* Whh   = (const float*)d_in[5];
    const float* bih   = (const float*)d_in[6];
    const float* bhh   = (const float*)d_in[7];
    const float* Wq    = (const float*)d_in[8];
    const float* bq    = (const float*)d_in[9];
    const float* Wk    = (const float*)d_in[10];
    const float* bk    = (const float*)d_in[11];
    const float* Wv    = (const float*)d_in[12];
    const float* bv    = (const float*)d_in[13];
    const float* Waoa  = (const float*)d_in[14];
    const float* baoa  = (const float*)d_in[15];
    const float* lng   = (const float*)d_in[16];
    const float* lnb   = (const float*)d_in[17];
    const float* predV = (const float*)d_in[18];
    const float* predg = (const float*)d_in[19];
    const float* predb = (const float*)d_in[20];
    float* out = (float*)d_out;

    char* ws = (char*)d_ws;
    size_t off = 0;
    auto alloc = [&](size_t bytes) -> char* {
        char* p = ws + off;
        off = (off + bytes + 255) & ~(size_t)255;
        return p;
    };

    bf16_t* Wcomb_bf = (bf16_t*)alloc((size_t)4 * H_ * KCOMB * 2);
    bf16_t* Wq_bf    = (bf16_t*)alloc((size_t)D_ * D_ * 2);
    bf16_t* Wk_bf    = (bf16_t*)alloc((size_t)D_ * D_ * 2);
    bf16_t* Wv_bf    = (bf16_t*)alloc((size_t)D_ * D_ * 2);
    bf16_t* Waoa_bf  = (bf16_t*)alloc((size_t)2 * D_ * 2 * D_ * 2);
    bf16_t* Wpred_bf = (bf16_t*)alloc((size_t)VPAD * H_ * 2);   // padded vocab rows
    float*  bsum     = (float*)alloc((size_t)4 * H_ * 4);
    bf16_t* enc_bf   = (bf16_t*)alloc((size_t)B_ * N_ * D_ * 2);
    float*  kp       = (float*)alloc((size_t)B_ * N_ * D_ * 4);
    float*  vp       = (float*)alloc((size_t)B_ * N_ * D_ * 4);
    float*  mf       = (float*)alloc((size_t)B_ * D_ * 4);
    float*  m_c      = (float*)alloc((size_t)B_ * H_ * 4);
    float*  ctx      = (float*)alloc((size_t)B_ * H_ * 4);
    bf16_t* xin      = (bf16_t*)alloc((size_t)B_ * KCOMB * 2);
    float*  gates    = (float*)alloc((size_t)B_ * 4 * H_ * 4);
    bf16_t* q_bf     = (bf16_t*)alloc((size_t)B_ * H_ * 2);
    float*  qp       = (float*)alloc((size_t)B_ * D_ * 4);
    bf16_t* zin      = (bf16_t*)alloc((size_t)B_ * 2 * D_ * 2);
    float*  z        = (float*)alloc((size_t)B_ * 2 * D_ * 4);
    bf16_t* ctx_bf   = (bf16_t*)alloc((size_t)B_ * H_ * 2);

    // ---- one-time prep (re-run every call: deterministic) ----
    // zero vocab padding rows first, then fill the real rows
    hipMemsetAsync(Wpred_bf + (size_t)V_ * H_, 0, (size_t)(VPAD - V_) * H_ * 2, stream);
    k_pack_comb<<<2048, 256, 0, stream>>>(Wih, Whh, Wcomb_bf);
    k_cvt_bf16<<<1024, 256, 0, stream>>>(Wq, Wq_bf, D_ * D_);
    k_cvt_bf16<<<1024, 256, 0, stream>>>(Wk, Wk_bf, D_ * D_);
    k_cvt_bf16<<<1024, 256, 0, stream>>>(Wv, Wv_bf, D_ * D_);
    k_cvt_bf16<<<2048, 256, 0, stream>>>(Waoa, Waoa_bf, 2 * D_ * 2 * D_);
    k_wnorm<<<V_, 256, 0, stream>>>(predV, predg, Wpred_bf);
    k_bias_sum<<<16, 256, 0, stream>>>(bih, bhh, bsum, 4 * H_);
    k_cvt_bf16<<<2048, 256, 0, stream>>>(enc, enc_bf, B_ * N_ * D_);
    k_meanfeat<<<(B_ * D_) / 256, 256, 0, stream>>>(enc, mf);
    hipMemsetAsync(m_c, 0, (size_t)B_ * H_ * 4, stream);
    hipMemsetAsync(ctx, 0, (size_t)B_ * H_ * 4, stream);
    hipMemsetAsync(xin, 0, (size_t)B_ * KCOMB * 2, stream);  // zero initial h cols

    // Npad must be a multiple of 512; Nreal masks the store epilogue.
    auto gemm = [&](const bf16_t* A, int lda, const bf16_t* W, int ldw,
                    const float* bias, float* C, int ldc,
                    int M, int Npad, int Nreal, int K, const int* L, int t) {
        dim3 grid(Npad / 512, M / 32);
        k_wmma_gemm_nt<<<grid, dim3(256), 0, stream>>>(A, lda, W, ldw, bias,
                                                       C, ldc, Nreal, K, L, t);
    };

    // ---- time-invariant K/V projections ----
    gemm(enc_bf, D_, Wk_bf, D_, bk, kp, D_, B_ * N_, D_, D_, D_, nullptr, 0);
    gemm(enc_bf, D_, Wv_bf, D_, bv, vp, D_, B_ * N_, D_, D_, D_, nullptr, 0);

    // ---- sequential decode steps ----
    for (int t = 0; t < T_; ++t) {
        k_embed_concat<<<(B_ * 2048) / 256, 256, 0, stream>>>(caps, embW, mf, ctx, xin, t);
        gemm(xin, KCOMB, Wcomb_bf, KCOMB, bsum, gates, 4 * H_,
             B_, 4 * H_, 4 * H_, KCOMB, nullptr, 0);
        k_lstm_ln<<<B_, 256, 0, stream>>>(gates, m_c, lng, lnb, xin, q_bf, zin);
        gemm(q_bf, H_, Wq_bf, D_, bq, qp, D_, B_, D_, D_, H_, nullptr, 0);
        k_attention<<<(B_ * NH_) / 256, 256, 0, stream>>>(qp, kp, vp, zin);
        gemm(zin, 2 * D_, Waoa_bf, 2 * D_, baoa, z, 2 * D_,
             B_, 2 * D_, 2 * D_, 2 * D_, nullptr, 0);
        k_glu<<<(B_ * H_) / 256, 256, 0, stream>>>(z, ctx, ctx_bf);
        gemm(ctx_bf, H_, Wpred_bf, H_, predb, out + (size_t)t * V_, T_ * V_,
             B_, VPAD, V_, H_, lens, t);
    }
}
